// DToxNet_61332132987630
// MI455X (gfx1250) — matrix operand order repaired and checked by reading, not compiled, not run on hardware
//
#include <hip/hip_runtime.h>

// ---------------------------------------------------------------------------
// Fused DToxNet for MI455X (gfx1250, wave32, WMMA).
//  - One workgroup (256 thr = 8 waves) per 16-row batch tile (512 blocks).
//  - All activations live in LDS (~273 KB of the WGP's 320 KB):
//      x tile 32KB | h0 128KB | h1 64KB | h2 32KB | h3 16KB | head scratch 1KB
//  - Per-node matmuls run on v_wmma_f32_16x16x32_f16 (f32 accumulate).
//  - Weights pre-converted once to f16 in B-fragment order in d_ws.
// HBM traffic ~= 32MB (x) + 16MB (outputs): memory-bound at ~2us ideal.
// ---------------------------------------------------------------------------

typedef __attribute__((ext_vector_type(16))) _Float16     v16h;
typedef __attribute__((ext_vector_type(8)))  float        v8f;
typedef __attribute__((ext_vector_type(4)))  unsigned int v4u;

union Frag16 {
    v16h     v;
    v4u      q[2];
    _Float16 f[16];
};

__device__ __forceinline__ float sigmoidf_(float x) {
    return 1.0f / (1.0f + __expf(-x));
}

#define B_TOT 8192
#define F_IN  1024
#define TPB   256
#define NWAVE (TPB / 32)

// LDS layout (bytes)
#define XS_OFF    0        // 16 x 1024 f16 = 32768
#define H0_OFF    32768    // 16 x 256 x 16 f16 = 131072
#define H1_OFF    163840   // 16 x 128 x 16 f16 = 65536
#define H2_OFF    229376   // 16 x  64 x 16 f16 = 32768
#define H3_OFF    262144   // 16 x  32 x 16 f16 = 16384
#define PART_OFF  278528   // 256 f32 = 1024
#define LDS_BYTES 279552

// d_ws layout (in halves): weight fragments, B-matrix order, zero padded.
// Per (node, kfrag): 32 lanes x 16 halves contiguous (1KB).
#define W0F_OFF 0          // 256*1*512
#define W1F_OFF 131072     // 128*2*512
#define W2F_OFF 262144     //  64*2*512
#define W3F_OFF 327680     //  32*2*512

// ---------------------------------------------------------------------------
// One-shot weight conversion: fp32 (n, m, 16) -> f16 WMMA-B fragments.
// dst[t]: t = ((node*nfrag + f)*32 + lane)*16 + e ; K = f*32 + (e<8?0:16) + (lane<16?0:8) + (e&7)
// ---------------------------------------------------------------------------
__global__ void __launch_bounds__(256)
cvt_weights(const float* __restrict__ W, _Float16* __restrict__ dst,
            int nfrag, int m_real, int total)
{
    int t = blockIdx.x * blockDim.x + threadIdx.x;
    if (t >= total) return;
    const int e    = t & 15;
    const int l    = (t >> 4) & 31;
    const int nf   = t >> 9;           // node*nfrag + f
    const int f    = nf % nfrag;
    const int node = nf / nfrag;
    const int k    = f * 32 + ((e < 8) ? 0 : 16) + ((l < 16) ? 0 : 8) + (e & 7);
    const int o    = l & 15;
    float v = (k < m_real) ? W[((size_t)node * m_real + k) * 16 + o] : 0.0f;
    dst[t] = (_Float16)v;
}

// ---------------------------------------------------------------------------
// Levels 1..3: per node, gather 4 parents (64 f16) from LDS, 2x WMMA K=32.
// ---------------------------------------------------------------------------
template <int NPREV, int NN>
__device__ __forceinline__ void level_ge1(
    int lane, int wave,
    const _Float16* __restrict__ hprev, _Float16* __restrict__ hcur,
    const int* __restrict__ idx, const _Float16* __restrict__ wfrag,
    const float* __restrict__ bias)
{
    const int m  = lane & 15;
    const int kb = (lane < 16) ? 0 : 8;   // A-layout K offset per half-wave
    const int ro = (lane < 16) ? 0 : 8;   // C/D row offset per half-wave
    const int o  = lane & 15;
    for (int n = wave; n < NN; n += NWAVE) {
        const int j0 = idx[n * 4 + 0], j1 = idx[n * 4 + 1];
        const int j2 = idx[n * 4 + 2], j3 = idx[n * 4 + 3];
        const _Float16* row = hprev + (size_t)m * NPREV * 16;
        Frag16 a0, a1, bf0, bf1;
        // A fragments: each 8-half group is one aligned 16B LDS read.
        a0.q[0] = *(const v4u*)(row + j0 * 16 + kb);
        a0.q[1] = *(const v4u*)(row + j1 * 16 + kb);
        a1.q[0] = *(const v4u*)(row + j2 * 16 + kb);
        a1.q[1] = *(const v4u*)(row + j3 * 16 + kb);
        // B fragments: coalesced 32B per lane from pre-packed ws.
        const v4u* wp0 = (const v4u*)(wfrag + ((size_t)(n * 2 + 0) * 512) + lane * 16);
        const v4u* wp1 = (const v4u*)(wfrag + ((size_t)(n * 2 + 1) * 512) + lane * 16);
        bf0.q[0] = wp0[0]; bf0.q[1] = wp0[1];
        bf1.q[0] = wp1[0]; bf1.q[1] = wp1[1];
        v8f c = {};
        c = __builtin_amdgcn_wmma_f32_16x16x32_f16(false, a0.v, false, bf0.v,
                                                   (short)0, c, false, false);
        c = __builtin_amdgcn_wmma_f32_16x16x32_f16(false, a1.v, false, bf1.v,
                                                   (short)0, c, false, false);
        const float bv = bias[n * 16 + o];
        _Float16* outp = hcur + n * 16 + o;
#pragma unroll
        for (int r = 0; r < 8; ++r) {
            float hv = c[r] + bv;
            hv = hv > 0.0f ? hv : 0.0f;
            outp[(size_t)(r + ro) * NN * 16] = (_Float16)hv;
        }
    }
}

// aux_l = sigmoid(sum_o h[b][n][o]*A[n][o] + a[n]) -> d_out region (B, NN)
template <int NN>
__device__ __forceinline__ void aux_level(
    int tid, int brow,
    const _Float16* __restrict__ h, const float* __restrict__ A,
    const float* __restrict__ aa, float* __restrict__ outp)
{
    for (int p = tid; p < 16 * NN; p += TPB) {
        const int mrow = p / NN;
        const int n    = p % NN;
        float s = aa[n];
        const _Float16* hp = h + (size_t)mrow * NN * 16 + n * 16;
        const float* Ap = A + n * 16;
#pragma unroll
        for (int o = 0; o < 16; ++o) s += (float)hp[o] * Ap[o];
        outp[(size_t)(brow + mrow) * NN + n] = sigmoidf_(s);
    }
}

__global__ void __launch_bounds__(TPB)
dtoxnet_fused(
    const float* __restrict__ x,
    const int* __restrict__ idx0, const int* __restrict__ idx1,
    const int* __restrict__ idx2, const int* __restrict__ idx3,
    const _Float16* __restrict__ wsf,
    const float* __restrict__ b0w, const float* __restrict__ A0, const float* __restrict__ a0,
    const float* __restrict__ b1w, const float* __restrict__ A1, const float* __restrict__ a1,
    const float* __restrict__ b2w, const float* __restrict__ A2, const float* __restrict__ a2,
    const float* __restrict__ b3w, const float* __restrict__ A3, const float* __restrict__ a3,
    const float* __restrict__ w_out, const float* __restrict__ b_out,
    float* __restrict__ dout)
{
    extern __shared__ char smem[];
    _Float16* xs = (_Float16*)(smem + XS_OFF);
    _Float16* h0 = (_Float16*)(smem + H0_OFF);
    _Float16* h1 = (_Float16*)(smem + H1_OFF);
    _Float16* h2 = (_Float16*)(smem + H2_OFF);
    _Float16* h3 = (_Float16*)(smem + H3_OFF);
    float*   part = (float*)(smem + PART_OFF);

    const int tid  = threadIdx.x;
    const int lane = tid & 31;
    const int wave = tid >> 5;
    const int brow = blockIdx.x * 16;

    // warm the (tiny, L2-resident) weight stream
    if (tid == 0) {
        __builtin_prefetch(wsf, 0, 0);
        __builtin_prefetch(w_out, 0, 0);
    }

    // stage the 16 x 1024 x-tile into LDS as f16 (coalesced)
    for (int i = tid; i < 16 * F_IN; i += TPB) {
        const int r = i >> 10, ccol = i & (F_IN - 1);
        xs[i] = (_Float16)x[(size_t)(brow + r) * F_IN + ccol];
    }
    __syncthreads();

    // ---- level 0: 256 nodes, K=8 padded to 32, one WMMA each ----
    {
        const int m  = lane & 15;
        const int ro = (lane < 16) ? 0 : 8;
        const int o  = lane & 15;
        const _Float16* w0f = wsf + W0F_OFF;
        for (int n = wave; n < 256; n += NWAVE) {
            Frag16 a;
            v4u z = {};
            a.q[0] = z; a.q[1] = z;
            if (lane < 16) {               // K=0..7 live in lanes 0-15 only
                const int* ip = idx0 + n * 8;
#pragma unroll
                for (int e = 0; e < 8; ++e) a.f[e] = xs[m * F_IN + ip[e]];
            }
            Frag16 b;
            const v4u* wp = (const v4u*)(w0f + (size_t)n * 512 + lane * 16);
            b.q[0] = wp[0]; b.q[1] = wp[1];
            v8f c = {};
            c = __builtin_amdgcn_wmma_f32_16x16x32_f16(false, a.v, false, b.v,
                                                       (short)0, c, false, false);
            const float bv = b0w[n * 16 + o];
            _Float16* outp = h0 + n * 16 + o;
#pragma unroll
            for (int r = 0; r < 8; ++r) {
                float hv = c[r] + bv;
                hv = hv > 0.0f ? hv : 0.0f;
                outp[(size_t)(r + ro) * 256 * 16] = (_Float16)hv;
            }
        }
    }
    __syncthreads();
    aux_level<256>(tid, brow, h0, A0, a0, dout + 8192);
    level_ge1<256, 128>(lane, wave, h0, h1, idx1, wsf + W1F_OFF, b1w);
    __syncthreads();
    aux_level<128>(tid, brow, h1, A1, a1, dout + 8192 + 8192 * 256);
    level_ge1<128, 64>(lane, wave, h1, h2, idx2, wsf + W2F_OFF, b2w);
    __syncthreads();
    aux_level<64>(tid, brow, h2, A2, a2, dout + 8192 + 8192 * (256 + 128));
    level_ge1<64, 32>(lane, wave, h2, h3, idx3, wsf + W3F_OFF, b3w);
    __syncthreads();
    aux_level<32>(tid, brow, h3, A3, a3, dout + 8192 + 8192 * (256 + 128 + 64));

    // ---- head: y = sigmoid(h3.flat(512) . w_out + b_out) ----
    {
        const int mrow = tid >> 4, q = tid & 15;
        const _Float16* hp = h3 + (size_t)mrow * 512;
        float s = 0.0f;
#pragma unroll
        for (int c2 = 0; c2 < 32; ++c2) {
            const int cc = q * 32 + c2;
            s += (float)hp[cc] * w_out[cc];
        }
        part[tid] = s;
    }
    __syncthreads();
    if (tid < 16) {
        float s = b_out[0];
#pragma unroll
        for (int q = 0; q < 16; ++q) s += part[tid * 16 + q];
        dout[brow + tid] = sigmoidf_(s);
    }
}

extern "C" void kernel_launch(void* const* d_in, const int* in_sizes, int n_in,
                              void* d_out, int out_size, void* d_ws, size_t ws_size,
                              hipStream_t stream)
{
    (void)in_sizes; (void)n_in; (void)out_size; (void)ws_size;
    const float* x    = (const float*)d_in[0];
    const int* idx0   = (const int*)d_in[1];
    const int* idx1   = (const int*)d_in[2];
    const int* idx2   = (const int*)d_in[3];
    const int* idx3   = (const int*)d_in[4];
    const float* W0   = (const float*)d_in[5];
    const float* b0w  = (const float*)d_in[6];
    const float* A0   = (const float*)d_in[7];
    const float* a0   = (const float*)d_in[8];
    const float* W1   = (const float*)d_in[9];
    const float* b1w  = (const float*)d_in[10];
    const float* A1   = (const float*)d_in[11];
    const float* a1   = (const float*)d_in[12];
    const float* W2   = (const float*)d_in[13];
    const float* b2w  = (const float*)d_in[14];
    const float* A2   = (const float*)d_in[15];
    const float* a2   = (const float*)d_in[16];
    const float* W3   = (const float*)d_in[17];
    const float* b3w  = (const float*)d_in[18];
    const float* A3   = (const float*)d_in[19];
    const float* a3   = (const float*)d_in[20];
    const float* wout = (const float*)d_in[21];
    const float* bout = (const float*)d_in[22];
    float* dout = (float*)d_out;

    _Float16* wsf = (_Float16*)d_ws;

    // pack weights -> f16 B fragments in ws (tiny, once per launch)
    cvt_weights<<<(256 * 1 * 512 + 255) / 256, 256, 0, stream>>>(
        W0, wsf + W0F_OFF, 1, 8, 256 * 1 * 512);
    cvt_weights<<<(128 * 2 * 512 + 255) / 256, 256, 0, stream>>>(
        W1, wsf + W1F_OFF, 2, 64, 128 * 2 * 512);
    cvt_weights<<<(64 * 2 * 512 + 255) / 256, 256, 0, stream>>>(
        W2, wsf + W2F_OFF, 2, 64, 64 * 2 * 512);
    cvt_weights<<<(32 * 2 * 512 + 255) / 256, 256, 0, stream>>>(
        W3, wsf + W3F_OFF, 2, 64, 32 * 2 * 512);

    (void)hipFuncSetAttribute((const void*)dtoxnet_fused,
                              hipFuncAttributeMaxDynamicSharedMemorySize,
                              LDS_BYTES);
    dtoxnet_fused<<<B_TOT / 16, TPB, LDS_BYTES, stream>>>(
        x, idx0, idx1, idx2, idx3, wsf,
        b0w, A0, a0, b1w, A1, a1, b2w, A2, a2, b3w, A3, a3,
        wout, bout, dout);
}
